// LSTMTagger_21148418966204
// MI455X (gfx1250) — matrix operand order repaired and checked by reading, not compiled, not run on hardware
//
#include <hip/hip_runtime.h>

// ---------------------------------------------------------------------------
// LSTM tagger for MI455X (gfx1250, wave32, WMMA).
// Pipeline: pack weights/embeddings to bf16 WMMA fragments -> 512 step kernels
// (v_wmma_f32_16x16x32_bf16 GEMM + LSTM cell, h kept in fragment order) ->
// tag projection + log_softmax (WMMA, TAG padded 50->64).
// GEMM loops are software-pipelined: next A/B fragments are loaded into
// registers while the current WMMA executes (compensates for low wave count
// on the latency-critical recurrence).
// ---------------------------------------------------------------------------

typedef __attribute__((ext_vector_type(16))) __bf16 bf16x16;
typedef __attribute__((ext_vector_type(8)))  float  f32x8;

// problem constants
constexpr int CE   = 300;     // embedding dim
constexpr int CH   = 1024;    // hidden
constexpr int CTAG = 50;
constexpr int CB   = 64;      // batch
constexpr int CT   = 512;     // seq len
constexpr int KE   = 320;     // embedding K padded to 10 k-tiles of 32
constexpr int KT_E = KE / 32; // 10
constexpr int KT_H = CH / 32; // 32
constexpr int FRAG = 512;     // bf16 elements per 16x32 fragment (32 lanes x 16)
constexpr int NT_G = (4 * CH) / 16; // 256 gate n-tiles

// workspace element counts
constexpr size_t SZ_WIH  = (size_t)NT_G * KT_E * FRAG;   // ushort
constexpr size_t SZ_WHH  = (size_t)NT_G * KT_H * FRAG;   // ushort
constexpr size_t SZ_WOUT = (size_t)4 * KT_H * FRAG;      // ushort
constexpr size_t SZ_BIAS = (size_t)4 * CH;               // float
constexpr size_t SZ_XE   = (size_t)CT * 4 * KT_E * FRAG; // ushort
constexpr size_t SZ_HF1  = (size_t)4 * KT_H * FRAG;      // one h buffer = 65536
constexpr size_t SZ_C    = (size_t)CB * CH;              // float
constexpr size_t SZ_LA   = (size_t)CT * SZ_HF1;          // ushort

static __device__ __forceinline__ unsigned short f2bf(float f) {
  unsigned u = __float_as_uint(f);
  return (unsigned short)((u + 0x7FFFu + ((u >> 16) & 1u)) >> 16);
}

static __device__ __forceinline__ bf16x16 ldfrag(const unsigned short* p) {
  return *(const bf16x16*)p;
}

static __device__ __forceinline__ f32x8 wmma_bf16(bf16x16 a, bf16x16 b, f32x8 c) {
  return __builtin_amdgcn_wmma_f32_16x16x32_bf16(false, a, false, b, (short)0, c,
                                                 false, false);
}

// A-fragment (16x32 bf16) intra-lane K index for (lane, halfword j):
//   lane<16 : K = {0..7, 16..23},  lane>=16 : K = {8..15, 24..31}
static __device__ __forceinline__ int a_k_in(int lane, int j) {
  return ((lane >> 4) & 1) * 8 + ((j >> 3) << 4) + (j & 7);
}

// ---------------------------------------------------------------------------
// Pack kernels: build bf16 fragment-linear weight / activation buffers.
// B-fragment (32x16): lane l holds column N=l&15, halfword j holds K=a_k_in(l,j).
// ---------------------------------------------------------------------------
__global__ void pack_wih_k(const float* __restrict__ w, unsigned short* __restrict__ p) {
  size_t e = (size_t)blockIdx.x * 256 + threadIdx.x;
  if (e >= SZ_WIH) return;
  int idx = (int)(e & (FRAG - 1));
  int lane = idx >> 4, j = idx & 15;
  int frag = (int)(e >> 9);
  int ktile = frag % KT_E, ntile = frag / KT_E;
  int n = ntile * 16 + (lane & 15), k = ktile * 32 + a_k_in(lane, j);
  float v = (k < CE) ? w[(size_t)n * CE + k] : 0.f;
  p[e] = f2bf(v);
}

__global__ void pack_whh_k(const float* __restrict__ w, unsigned short* __restrict__ p) {
  size_t e = (size_t)blockIdx.x * 256 + threadIdx.x;
  if (e >= SZ_WHH) return;
  int idx = (int)(e & (FRAG - 1));
  int lane = idx >> 4, j = idx & 15;
  int frag = (int)(e >> 9);
  int ktile = frag & (KT_H - 1), ntile = frag >> 5;
  int n = ntile * 16 + (lane & 15), k = ktile * 32 + a_k_in(lane, j);
  p[e] = f2bf(w[(size_t)n * CH + k]);
}

__global__ void pack_wout_k(const float* __restrict__ w, unsigned short* __restrict__ p) {
  size_t e = (size_t)blockIdx.x * 256 + threadIdx.x;
  if (e >= SZ_WOUT) return;
  int idx = (int)(e & (FRAG - 1));
  int lane = idx >> 4, j = idx & 15;
  int frag = (int)(e >> 9);
  int ktile = frag & (KT_H - 1), ntile = frag >> 5;
  int n = ntile * 16 + (lane & 15), k = ktile * 32 + a_k_in(lane, j);
  float v = (n < CTAG) ? w[(size_t)n * CH + k] : 0.f;
  p[e] = f2bf(v);
}

__global__ void pack_bias_k(const float* __restrict__ bi, const float* __restrict__ bh,
                            float* __restrict__ p) {
  int e = blockIdx.x * 256 + threadIdx.x;
  if (e < 4 * CH) p[e] = bi[e] + bh[e];
}

// Embedding gather directly into A-fragment order: Xe[t][mtile][ktile][lane][16]
__global__ void pack_x_k(const int* __restrict__ sent, const float* __restrict__ emb,
                         unsigned short* __restrict__ p) {
  size_t e = (size_t)blockIdx.x * 256 + threadIdx.x;
  if (e >= SZ_XE) return;
  int idx = (int)(e & (FRAG - 1));
  int lane = idx >> 4, j = idx & 15;
  int frag = (int)(e >> 9);
  int ktile = frag % KT_E;
  int rem = frag / KT_E;
  int mtile = rem & 3;
  int t = rem >> 2;
  int b = mtile * 16 + (lane & 15);
  int k = ktile * 32 + a_k_in(lane, j);
  float v = 0.f;
  if (k < CE) {
    int tok = sent[(size_t)b * CT + t];  // sentence is (B, T)
    v = emb[(size_t)tok * CE + k];
  }
  p[e] = f2bf(v);
}

__global__ void init_state_k(unsigned short* __restrict__ h0, float* __restrict__ c) {
  int e = blockIdx.x * 256 + threadIdx.x;
  if (e < (int)SZ_HF1) { h0[e] = 0; c[e] = 0.f; }
}

// ---------------------------------------------------------------------------
// One LSTM time step. Grid: 16 workgroups (64 hidden units each), 256 threads.
// Wave w: M-tile = w&3 (16 batch rows), 8 gate N-tiles = (w>>2)*8 ..+7.
// gates staged in LDS so the cell sees i/f/g/o per hidden unit; h_new written
// back in A-fragment order for the next step (ping-pong).
// ---------------------------------------------------------------------------
__global__ __launch_bounds__(256) void lstm_step_k(
    const unsigned short* __restrict__ Xe,    // [T][4][KT_E][FRAG]
    const unsigned short* __restrict__ WihP,  // [NT_G][KT_E][FRAG]
    const unsigned short* __restrict__ WhhP,  // [NT_G][KT_H][FRAG]
    const float* __restrict__ bias,           // [4H]
    const int* __restrict__ lengths,          // [B]
    unsigned short* __restrict__ Hfrag,       // [2][4][KT_H][FRAG]
    float* __restrict__ Cst,                  // [B][H]
    unsigned short* __restrict__ LstmA,       // [T][4][KT_H][FRAG]
    int t) {
  __shared__ float gl[4][CB][64];

  const int tid = threadIdx.x;
  const int lane = tid & 31, wave = tid >> 5;
  const int mtile = wave & 3, ngrp = wave >> 2;
  const int hs = blockIdx.x;  // hidden slice [hs*64, hs*64+64)

  const f32x8 z = {0.f, 0.f, 0.f, 0.f, 0.f, 0.f, 0.f, 0.f};
  f32x8 acc[8];
#pragma unroll
  for (int u = 0; u < 8; ++u) acc[u] = z;

  const unsigned short* bhh[8];
  const unsigned short* bih[8];
  int gArr[8], nt4Arr[8];
#pragma unroll
  for (int u = 0; u < 8; ++u) {
    int nl = ngrp * 8 + u;
    int g = nl >> 2, nt4 = nl & 3;
    int ntg = g * 64 + hs * 4 + nt4;  // global gate n-tile
    bhh[u] = WhhP + (size_t)ntg * KT_H * FRAG + lane * 16;
    bih[u] = WihP + (size_t)ntg * KT_E * FRAG + lane * 16;
    gArr[u] = g; nt4Arr[u] = nt4;
  }

  // ---- h @ W_hh^T (K = 1024), software-pipelined ----
  {
    const unsigned short* A =
        Hfrag + ((size_t)(t & 1) * 4 + mtile) * KT_H * FRAG + lane * 16;
    bf16x16 a_cur = ldfrag(A);
    bf16x16 b_cur = ldfrag(bhh[0]);
#pragma unroll 1
    for (int kt = 0; kt < KT_H - 1; ++kt) {
      bf16x16 a_nxt = ldfrag(A + (size_t)(kt + 1) * FRAG);
#pragma unroll
      for (int u = 0; u < 8; ++u) {
        bf16x16 b_nxt = (u < 7) ? ldfrag(bhh[u + 1] + (size_t)kt * FRAG)
                                : ldfrag(bhh[0] + (size_t)(kt + 1) * FRAG);
        acc[u] = wmma_bf16(a_cur, b_cur, acc[u]);
        b_cur = b_nxt;
      }
      a_cur = a_nxt;
    }
#pragma unroll
    for (int u = 0; u < 8; ++u) {  // epilogue kt = KT_H-1
      bf16x16 b_nxt =
          (u < 7) ? ldfrag(bhh[u + 1] + (size_t)(KT_H - 1) * FRAG) : b_cur;
      acc[u] = wmma_bf16(a_cur, b_cur, acc[u]);
      b_cur = b_nxt;
    }
  }
  // ---- x_t @ W_ih^T (K = 320, zero-padded from 300), software-pipelined ----
  {
    const unsigned short* A =
        Xe + ((size_t)t * 4 + mtile) * KT_E * FRAG + lane * 16;
    bf16x16 a_cur = ldfrag(A);
    bf16x16 b_cur = ldfrag(bih[0]);
#pragma unroll 1
    for (int kt = 0; kt < KT_E - 1; ++kt) {
      bf16x16 a_nxt = ldfrag(A + (size_t)(kt + 1) * FRAG);
#pragma unroll
      for (int u = 0; u < 8; ++u) {
        bf16x16 b_nxt = (u < 7) ? ldfrag(bih[u + 1] + (size_t)kt * FRAG)
                                : ldfrag(bih[0] + (size_t)(kt + 1) * FRAG);
        acc[u] = wmma_bf16(a_cur, b_cur, acc[u]);
        b_cur = b_nxt;
      }
      a_cur = a_nxt;
    }
#pragma unroll
    for (int u = 0; u < 8; ++u) {  // epilogue kt = KT_E-1
      bf16x16 b_nxt =
          (u < 7) ? ldfrag(bih[u + 1] + (size_t)(KT_E - 1) * FRAG) : b_cur;
      acc[u] = wmma_bf16(a_cur, b_cur, acc[u]);
      b_cur = b_nxt;
    }
  }

  // C layout: VGPR r, lane l -> (M = (l>>4)*8 + r, N = l&15)
  const int mrow = mtile * 16 + ((lane >> 4) << 3);
  const int ncol = lane & 15;
#pragma unroll
  for (int u = 0; u < 8; ++u) {
    int g = gArr[u], nt4 = nt4Arr[u];
    int nloc = nt4 * 16 + ncol;
    float bv = bias[g * CH + hs * 64 + nloc];
#pragma unroll
    for (int r = 0; r < 8; ++r) gl[g][mrow + r][nloc] = acc[u][r] + bv;
  }
  __syncthreads();

  // Elementwise LSTM cell on the 64(batch) x 64(hidden) slice.
#pragma unroll 1
  for (int it = 0; it < 16; ++it) {
    int e = it * 256 + tid;
    int b = e >> 6, n = e & 63;
    int nglob = hs * 64 + n;
    float iv = gl[0][b][n], fv = gl[1][b][n], gv = gl[2][b][n], ov = gl[3][b][n];
    float si = 1.f / (1.f + __expf(-iv));
    float sf = 1.f / (1.f + __expf(-fv));
    float gg = tanhf(gv);
    float so = 1.f / (1.f + __expf(-ov));
    float cold = Cst[(size_t)b * CH + nglob];
    float cnew = sf * cold + si * gg;
    float hnew = so * tanhf(cnew);

    // A-fragment coords for next step: row = b, K = nglob
    int mt = b >> 4, m_in = b & 15;
    int kt = nglob >> 5, k_in = nglob & 31;
    int fl = m_in | ((k_in & 8) << 1);
    int fj = ((k_in >> 4) << 3) | (k_in & 7);
    size_t fidx = (((size_t)mt * KT_H + kt) * 32 + fl) * 16 + fj;

    bool live = t < lengths[b];  // lengths sorted desc -> mask monotone in t
    unsigned short hb, ob;
    if (live) {
      Cst[(size_t)b * CH + nglob] = cnew;
      hb = f2bf(hnew);
      ob = hb;
    } else {
      hb = Hfrag[(size_t)(t & 1) * SZ_HF1 + fidx];  // carry frozen h
      ob = 0;                                       // out = 0 for padded rows
    }
    Hfrag[(size_t)((t + 1) & 1) * SZ_HF1 + fidx] = hb;
    LstmA[(size_t)t * SZ_HF1 + fidx] = ob;
  }
}

// ---------------------------------------------------------------------------
// Tag projection + log_softmax. One wave per 16 (t,b) rows; 4 WMMA N-tiles
// cover TAG=50 padded to 64; softmax over 50 valid columns via LDS.
// ---------------------------------------------------------------------------
__global__ __launch_bounds__(32) void tag_k(
    const unsigned short* __restrict__ LstmA,  // [T*4][KT_H][FRAG]
    const unsigned short* __restrict__ WoutP,  // [4][KT_H][FRAG]
    const float* __restrict__ b_out,
    float* __restrict__ out) {
  __shared__ float tl[16][64];
  const int lane = threadIdx.x;
  const int mg = blockIdx.x;  // global M-tile = t*4 + mtile, 0..2047
  const unsigned short* Ab = LstmA + (size_t)mg * KT_H * FRAG + lane * 16;

  const unsigned short* wob[4];
#pragma unroll
  for (int nt = 0; nt < 4; ++nt)
    wob[nt] = WoutP + (size_t)nt * KT_H * FRAG + lane * 16;

  const f32x8 z = {0.f, 0.f, 0.f, 0.f, 0.f, 0.f, 0.f, 0.f};
  f32x8 acc[4] = {z, z, z, z};

  bf16x16 a_cur = ldfrag(Ab);
  bf16x16 b_cur = ldfrag(wob[0]);
#pragma unroll 1
  for (int kt = 0; kt < KT_H - 1; ++kt) {
    bf16x16 a_nxt = ldfrag(Ab + (size_t)(kt + 1) * FRAG);
#pragma unroll
    for (int nt = 0; nt < 4; ++nt) {
      bf16x16 b_nxt = (nt < 3) ? ldfrag(wob[nt + 1] + (size_t)kt * FRAG)
                               : ldfrag(wob[0] + (size_t)(kt + 1) * FRAG);
      acc[nt] = wmma_bf16(a_cur, b_cur, acc[nt]);
      b_cur = b_nxt;
    }
    a_cur = a_nxt;
  }
#pragma unroll
  for (int nt = 0; nt < 4; ++nt) {  // epilogue kt = KT_H-1
    bf16x16 b_nxt =
        (nt < 3) ? ldfrag(wob[nt + 1] + (size_t)(KT_H - 1) * FRAG) : b_cur;
    acc[nt] = wmma_bf16(a_cur, b_cur, acc[nt]);
    b_cur = b_nxt;
  }

  const int mrow = (lane >> 4) << 3, ncol = lane & 15;
#pragma unroll
  for (int nt = 0; nt < 4; ++nt)
#pragma unroll
    for (int r = 0; r < 8; ++r) tl[mrow + r][nt * 16 + ncol] = acc[nt][r];
  __syncthreads();

  if (lane < 16) {
    int row = lane;
    int t = mg >> 2, b = (mg & 3) * 16 + row;
    size_t ob = ((size_t)t * CB + b) * CTAG;
    float mx = -3.4e38f;
    for (int k = 0; k < CTAG; ++k) mx = fmaxf(mx, tl[row][k] + b_out[k]);
    float s = 0.f;
    for (int k = 0; k < CTAG; ++k) s += __expf(tl[row][k] + b_out[k] - mx);
    float lz = mx + __logf(s);
    for (int k = 0; k < CTAG; ++k) out[ob + k] = (tl[row][k] + b_out[k]) - lz;
  }
}

// ---------------------------------------------------------------------------
extern "C" void kernel_launch(void* const* d_in, const int* in_sizes, int n_in,
                              void* d_out, int out_size, void* d_ws, size_t ws_size,
                              hipStream_t stream) {
  const int*   sentence = (const int*)  d_in[0];
  const int*   lengths  = (const int*)  d_in[1];
  const float* emb      = (const float*)d_in[2];
  const float* W_ih     = (const float*)d_in[3];
  const float* W_hh     = (const float*)d_in[4];
  const float* b_ih     = (const float*)d_in[5];
  const float* b_hh     = (const float*)d_in[6];
  const float* W_out    = (const float*)d_in[7];
  const float* b_out    = (const float*)d_in[8];
  float* out = (float*)d_out;
  (void)in_sizes; (void)n_in; (void)out_size; (void)ws_size;

  char* ws = (char*)d_ws;
  size_t off = 0;
  auto take = [&](size_t bytes) -> char* {
    char* p = ws + off;
    off = (off + bytes + 255) & ~(size_t)255;
    return p;
  };
  unsigned short* WihP  = (unsigned short*)take(SZ_WIH  * 2);
  unsigned short* WhhP  = (unsigned short*)take(SZ_WHH  * 2);
  unsigned short* WoutP = (unsigned short*)take(SZ_WOUT * 2);
  float*          biasB = (float*)         take(SZ_BIAS * 4);
  unsigned short* Xe    = (unsigned short*)take(SZ_XE   * 2);
  unsigned short* Hf    = (unsigned short*)take(2 * SZ_HF1 * 2);
  float*          Cst   = (float*)         take(SZ_C    * 4);
  unsigned short* La    = (unsigned short*)take(SZ_LA   * 2);

  pack_wih_k <<<(unsigned)((SZ_WIH  + 255) / 256), 256, 0, stream>>>(W_ih, WihP);
  pack_whh_k <<<(unsigned)((SZ_WHH  + 255) / 256), 256, 0, stream>>>(W_hh, WhhP);
  pack_wout_k<<<(unsigned)((SZ_WOUT + 255) / 256), 256, 0, stream>>>(W_out, WoutP);
  pack_bias_k<<<(4 * CH + 255) / 256, 256, 0, stream>>>(b_ih, b_hh, biasB);
  pack_x_k   <<<(unsigned)((SZ_XE   + 255) / 256), 256, 0, stream>>>(sentence, emb, Xe);
  init_state_k<<<((int)SZ_HF1 + 255) / 256, 256, 0, stream>>>(Hf, Cst);

  for (int t = 0; t < CT; ++t)
    lstm_step_k<<<16, 256, 0, stream>>>(Xe, WihP, WhhP, biasB, lengths, Hf, Cst, La, t);

  tag_k<<<2048, 32, 0, stream>>>(La, WoutP, b_out, out);
}
